// Attention_29618094473452
// MI455X (gfx1250) — matrix operand order repaired and checked by reading, not compiled, not run on hardware
//
#include <hip/hip_runtime.h>
#include <cstdint>
#include <cstddef>

// Problem constants (from reference)
constexpr int kTT  = 1024;
constexpr int kTS  = 1024;
constexpr int kB   = 32;
constexpr int kH   = 512;
constexpr int kOUT = 512;

// ---------- CDNA5 WMMA plumbing (wave32, 16x16x32 bf16 -> f32) ----------
typedef __attribute__((ext_vector_type(16))) __bf16    v16bf;
typedef __attribute__((ext_vector_type(2)))  __bf16    v2bf;
typedef __attribute__((ext_vector_type(8)))  float     v8f;
typedef __attribute__((ext_vector_type(4)))  uint32_t  uint4v;  // trivial 16B vector
typedef __attribute__((ext_vector_type(4)))  float     flt4v;

union FragB {           // one WMMA A or B operand (16 bf16 per lane = 8 VGPRs)
    v16bf    v;
    uint32_t u[8];
    uint4v   q[2];
};
union FragC {           // one WMMA C/D accumulator (8 f32 per lane)
    v8f   v;
    float f[8];
};

__device__ __forceinline__ v8f wmma_bf16(const FragB& a, const FragB& b, v8f c) {
    // (neg_a, A, neg_b, B, c_mod, C, reuse_a, reuse_b)
    return __builtin_amdgcn_wmma_f32_16x16x32_bf16(false, a.v, false, b.v,
                                                   (short)0, c, false, false);
}

// ---- f32 -> bf16 (RNE). Prefer HW packed converter v_cvt_pk_bf16_f32. ----
#if defined(__has_builtin)
#  if __has_builtin(__builtin_amdgcn_cvt_pk_bf16_f32)
#    define ATTN_HAVE_CVT_PK_BF16 1
#  endif
#endif
#ifndef ATTN_HAVE_CVT_PK_BF16
#  define ATTN_USE_CVT_ASM 1      // set to 0 if assembler rejects v_cvt_pk_bf16_f32
#endif

__device__ __forceinline__ uint32_t pack2bf(float a, float b) {
#if defined(ATTN_HAVE_CVT_PK_BF16)
    union { v2bf v; uint32_t u; } r;
    r.v = __builtin_amdgcn_cvt_pk_bf16_f32(a, b);
    return r.u;
#elif defined(ATTN_USE_CVT_ASM) && ATTN_USE_CVT_ASM
    uint32_t r;
    asm("v_cvt_pk_bf16_f32 %0, %1, %2" : "=v"(r) : "v"(a), "v"(b));
    return r;
#else
    uint32_t ua = __float_as_uint(a);
    uint32_t ub = __float_as_uint(b);
    ua += 0x7fffu + ((ua >> 16) & 1u);
    ub += 0x7fffu + ((ub >> 16) & 1u);
    return (ua >> 16) | (ub & 0xffff0000u);
#endif
}
__device__ __forceinline__ uint16_t f2bf(float x) {
    return (uint16_t)(pack2bf(x, 0.f) & 0xffffu);
}
// Load 8 contiguous f32, convert to 8 bf16 packed into 4 dwords
__device__ __forceinline__ void load_cvt8(const float* __restrict__ p, uint32_t* u) {
    flt4v f0 = *(const flt4v*)p;
    flt4v f1 = *(const flt4v*)(p + 4);
    u[0] = pack2bf(f0.x, f0.y);
    u[1] = pack2bf(f0.z, f0.w);
    u[2] = pack2bf(f1.x, f1.y);
    u[3] = pack2bf(f1.z, f1.w);
}

// A-fragment (16-bit A 16x32 layout): lane holds K {base+0..7, base+16..23},
// base = k0 + 8*(lane>=16).  src must be 16B aligned bf16.
__device__ __forceinline__ void load_fragA_bf16(const uint16_t* src, FragB& A) {
    const uint4v* aq = (const uint4v*)src;
    A.q[0] = aq[0];     // K +0..7
    A.q[1] = aq[2];     // K +16..23
}
// B-fragment: lane holds K {base+0..7, base+16..23}, base = k0 + 8*(lane>=16).
__device__ __forceinline__ void load_fragB_bf16(const uint16_t* src, FragB& B) {
    const uint4v* bq = (const uint4v*)src;
    B.q[0] = bq[0];
    B.q[1] = bq[2];
}

// ---------- Kernel 1: weights -> transposed bf16 (WaT[n][k], WcT[n][k]) ----------
__global__ void __launch_bounds__(256) wtrans_kernel(
    const float* __restrict__ Wa, const float* __restrict__ Wc,
    uint16_t* __restrict__ WaT, uint16_t* __restrict__ WcT)
{
    const size_t idx = (size_t)blockIdx.x * 256 + threadIdx.x;
    if (idx < (size_t)kH * kH) {                        // WaT: 512x512
        const int k = (int)(idx & (kH - 1));
        const int n = (int)(idx >> 9);
        WaT[idx] = f2bf(Wa[(size_t)k * kH + n]);
    } else {                                            // WcT: 512 x 1024
        const size_t j = idx - (size_t)kH * kH;
        const int k = (int)(j & (2 * kH - 1));
        const int n = (int)(j >> 10);
        WcT[j] = f2bf(Wc[(size_t)k * kOUT + n]);
    }
}

// ---------- Kernel 2: hsT[i][n][k] = bf16(hs[k][i][n])  (K-contiguous B operand) ----------
__global__ void __launch_bounds__(256) hst_kernel(
    const float* __restrict__ hs, uint16_t* __restrict__ hsT)
{
    const size_t idx = (size_t)blockIdx.x * 256 + threadIdx.x;   // 32*1024*512
    const int n = (int)(idx & (kH - 1));
    const size_t rem = idx >> 9;
    const int k = (int)(rem & (kTS - 1));
    const int i = (int)(rem >> 10);
    hsT[((size_t)i * kH + n) * kTS + k] =
        f2bf(hs[((size_t)k * kB + i) * kH + n]);
}

// ---------- Kernel 3: proj[i][l][n] = bf16( hs[l,i,:] @ Wa )  (WMMA GEMM) ----------
// WG = 4 waves; wave tile 16(M) x 64(N); WG tile 16 x 256.
__global__ void __launch_bounds__(128) proj_gemm_kernel(
    const float* __restrict__ hs, const uint16_t* __restrict__ WaT,
    uint16_t* __restrict__ proj)
{
    const int i    = blockIdx.x >> 6;          // batch
    const int l0   = (blockIdx.x & 63) << 4;   // ts-row tile
    const int nb   = blockIdx.y * 256;
    const int wave = threadIdx.x >> 5;
    const int lane = threadIdx.x & 31;
    const int lm   = lane & 15;
    const int hi   = lane >> 4;

    const float* arow = hs + ((size_t)(l0 + lm) * kB + i) * kH;
    FragC acc[4];
    for (int t2 = 0; t2 < 4; ++t2)
        for (int z = 0; z < 8; ++z) acc[t2].f[z] = 0.f;

    for (int k0 = 0; k0 < kH; k0 += 32) {
        FragB A;
        load_cvt8(arow + k0 + hi * 8,      A.u);      // K +0..7
        load_cvt8(arow + k0 + hi * 8 + 16, A.u + 4);  // K +16..23
        for (int t2 = 0; t2 < 4; ++t2) {
            const int n = nb + wave * 64 + t2 * 16 + lm;
            FragB Bf;
            load_fragB_bf16(WaT + (size_t)n * kH + k0 + hi * 8, Bf);
            acc[t2].v = wmma_bf16(A, Bf, acc[t2].v);
        }
    }
    for (int t2 = 0; t2 < 4; ++t2) {
        const int n = nb + wave * 64 + t2 * 16 + lm;
        for (int v = 0; v < 8; ++v) {
            const int m = v + 8 * hi;
            proj[((size_t)i * kTS + l0 + m) * kH + n] = f2bf(acc[t2].f[v]);
        }
    }
}

// ---------- Kernel 4: fused attention (score GEMM + softmax + context GEMM) ----------
// Grid: (TT/16, B). WG = 128 threads = 4 waves. Dynamic LDS ~114 KB (CDNA5: 320 KB/WGP).
constexpr int kSmemA    = 0;                       // 16x512 bf16  (16 KB)   staged ht tile
constexpr int kSmemS    = 16384;                   // 16x1024 f32  (64 KB)   score stripe
constexpr int kSmemAw   = 16384 + 65536;           // 16x1024 bf16 (32 KB)   softmaxed a
constexpr int kSmemMask = kSmemAw + 32768;         // 1024 B
constexpr int kSmemRed  = kSmemMask + 1024;        // 16x8 f32
constexpr int kSmemRMax = kSmemRed + 512;          // 16 f32
constexpr int kSmemRInv = kSmemRMax + 64;          // 16 f32
constexpr int kSmemSize = kSmemRInv + 64;          // 116352 bytes

__global__ void __launch_bounds__(128) attn_kernel(
    const float* __restrict__ ht, const long long* __restrict__ source,
    const uint16_t* __restrict__ proj, const uint16_t* __restrict__ hsT,
    uint16_t* __restrict__ cbf)
{
    extern __shared__ char smem[];
    uint16_t*      sA    = (uint16_t*)(smem + kSmemA);
    float*         sS    = (float*)   (smem + kSmemS);
    uint16_t*      sab   = (uint16_t*)(smem + kSmemAw);
    unsigned char* smask = (unsigned char*)(smem + kSmemMask);
    float*         sred  = (float*)(smem + kSmemRed);
    float*         srmax = (float*)(smem + kSmemRMax);
    float*         srinv = (float*)(smem + kSmemRInv);

    const int j0   = blockIdx.x * 16;
    const int i    = blockIdx.y;
    const int t    = threadIdx.x;
    const int wave = t >> 5;
    const int lane = t & 31;
    const int lm   = lane & 15;
    const int hi   = lane >> 4;

    // ---- Phase 0: stage ht tile (16 x 512) as bf16 in LDS; build mask ----
    uint32_t* sA32 = (uint32_t*)sA;
    for (int p = t; p < 16 * 256; p += 128) {              // f32 pairs
        const int row = p >> 8;
        const int c2  = (p & 255) << 1;
        const float* src = ht + ((size_t)(j0 + row) * kB + i) * kH + c2;
        sA32[p] = pack2bf(src[0], src[1]);
    }
    for (int k = t; k < kTS; k += 128)
        smask[k] = (source[(size_t)k * kB + i] == 0) ? 1 : 0;
    __syncthreads();

    // ---- Phase 1: S[16 x 1024] = ht_tile @ proj^T; waves split TS,
    //      4 N-tiles per K-step share one A fragment (4 WMMA chains in flight) ----
    for (int tg = 0; tg < 4; ++tg) {
        const int tsg = wave * 256 + tg * 64;
        FragC acc[4];
        for (int t2 = 0; t2 < 4; ++t2)
            for (int z = 0; z < 8; ++z) acc[t2].f[z] = 0.f;
        for (int k0 = 0; k0 < kH; k0 += 32) {
            FragB A;
            load_fragA_bf16(sA + lm * kH + k0 + hi * 8, A);
            for (int t2 = 0; t2 < 4; ++t2) {
                const int ts0 = tsg + t2 * 16;
                FragB Bf;
                load_fragB_bf16(proj + ((size_t)i * kTS + ts0 + lm) * kH + k0 + hi * 8, Bf);
                acc[t2].v = wmma_bf16(A, Bf, acc[t2].v);
            }
        }
        for (int t2 = 0; t2 < 4; ++t2) {
            const int ts0 = tsg + t2 * 16;
            for (int v = 0; v < 8; ++v)
                sS[(v + 8 * hi) * kTS + ts0 + lm] = acc[t2].f[v];
        }
    }
    __syncthreads();

    // ---- Phase 2: exact softmax with post-exp masking (matches reference) ----
    const int r   = t >> 3;       // row 0..15
    const int seg = t & 7;        // 128-col segment
    const int c0  = seg * 128;
    float mx = -3.402823466e38f;
    for (int c = c0; c < c0 + 128; ++c) mx = fmaxf(mx, sS[r * kTS + c]);
    sred[r * 8 + seg] = mx;
    __syncthreads();
    if (seg == 0) {
        float m2 = sred[r * 8];
        for (int q = 1; q < 8; ++q) m2 = fmaxf(m2, sred[r * 8 + q]);
        srmax[r] = m2;
    }
    __syncthreads();
    const float rowm = srmax[r];
    float sum = 0.f;
    for (int c = c0; c < c0 + 128; ++c) {
        const float e = smask[c] ? 0.f : __expf(sS[r * kTS + c] - rowm);
        sab[r * kTS + c] = f2bf(e);
        sum += e;
    }
    sred[r * 8 + seg] = sum;
    __syncthreads();
    if (seg == 0) {
        float s2 = 0.f;
        for (int q = 0; q < 8; ++q) s2 += sred[r * 8 + q];
        srinv[r] = 1.0f / s2;
    }
    __syncthreads();

    // ---- Phase 3: c = (a/rowsum) @ hs; waves split H (128 cols each),
    //      4 N-tiles per K-step share one A fragment ----
    for (int tg = 0; tg < 2; ++tg) {
        FragC acc[4];
        for (int t2 = 0; t2 < 4; ++t2)
            for (int z = 0; z < 8; ++z) acc[t2].f[z] = 0.f;
        for (int k0 = 0; k0 < kTS; k0 += 32) {
            FragB A;
            load_fragA_bf16(sab + lm * kTS + k0 + hi * 8, A);
            for (int t2 = 0; t2 < 4; ++t2) {
                const int n = wave * 128 + tg * 64 + t2 * 16 + lm;
                FragB Bf;
                load_fragB_bf16(hsT + ((size_t)i * kH + n) * kTS + k0 + hi * 8, Bf);
                acc[t2].v = wmma_bf16(A, Bf, acc[t2].v);
            }
        }
        for (int t2 = 0; t2 < 4; ++t2) {
            const int n = wave * 128 + tg * 64 + t2 * 16 + lm;
            for (int v = 0; v < 8; ++v) {
                const int m = v + 8 * hi;
                const float val = acc[t2].f[v] * srinv[m];
                cbf[((size_t)(j0 + m) * kB + i) * kH + n] = f2bf(val);
            }
        }
    }
}

// ---------- Kernel 5: out = tanh([c, ht] @ Wc + b)  (WMMA GEMM, fused epilogue) ----------
__global__ void __launch_bounds__(128) out_gemm_kernel(
    const float* __restrict__ ht, const uint16_t* __restrict__ cbf,
    const uint16_t* __restrict__ WcT, const float* __restrict__ bias,
    float* __restrict__ out)
{
    const int r0   = blockIdx.x * 16;          // flat row over TT*B
    const int nb   = blockIdx.y * 256;
    const int wave = threadIdx.x >> 5;
    const int lane = threadIdx.x & 31;
    const int lm   = lane & 15;
    const int hi   = lane >> 4;

    const uint16_t* crow = cbf + (size_t)(r0 + lm) * kH;
    const float*    hrow = ht  + (size_t)(r0 + lm) * kH;
    FragC acc[4];
    for (int t2 = 0; t2 < 4; ++t2)
        for (int z = 0; z < 8; ++z) acc[t2].f[z] = 0.f;

    for (int k0 = 0; k0 < 2 * kH; k0 += 32) {
        FragB A;
        if (k0 < kH) {                      // first half of K: c (already bf16)
            load_fragA_bf16(crow + k0 + hi * 8, A);
        } else {                            // second half of K: ht (f32 -> bf16)
            load_cvt8(hrow + (k0 - kH) + hi * 8,      A.u);
            load_cvt8(hrow + (k0 - kH) + hi * 8 + 16, A.u + 4);
        }
        for (int t2 = 0; t2 < 4; ++t2) {
            const int n = nb + wave * 64 + t2 * 16 + lm;
            FragB Bf;
            load_fragB_bf16(WcT + (size_t)n * (2 * kH) + k0 + hi * 8, Bf);
            acc[t2].v = wmma_bf16(A, Bf, acc[t2].v);
        }
    }
    for (int t2 = 0; t2 < 4; ++t2) {
        const int n  = nb + wave * 64 + t2 * 16 + lm;
        const float bn = bias[n];
        for (int v = 0; v < 8; ++v) {
            const int m = v + 8 * hi;
            out[(size_t)(r0 + m) * kOUT + n] = tanhf(acc[t2].f[v] + bn);
        }
    }
}

// ---------- Host launcher ----------
extern "C" void kernel_launch(void* const* d_in, const int* in_sizes, int n_in,
                              void* d_out, int out_size, void* d_ws, size_t ws_size,
                              hipStream_t stream)
{
    (void)in_sizes; (void)n_in; (void)out_size; (void)ws_size;
    const float*     ht     = (const float*)d_in[0];
    const float*     hs     = (const float*)d_in[1];
    const long long* source = (const long long*)d_in[2];
    const float*     Wa     = (const float*)d_in[3];
    const float*     Wc     = (const float*)d_in[4];
    const float*     bias   = (const float*)d_in[5];
    float*           out    = (float*)d_out;

    // Workspace layout (all 16B aligned): ~98 MB total
    char* ws = (char*)d_ws;
    constexpr size_t szWaT  = (size_t)kH * kH * 2;            // 512 KB
    constexpr size_t szWcT  = (size_t)kOUT * 2 * kH * 2;      // 1 MB
    constexpr size_t szProj = (size_t)kB * kTS * kH * 2;      // 32 MB
    constexpr size_t szHsT  = (size_t)kB * kH * kTS * 2;      // 32 MB
    uint16_t* WaT  = (uint16_t*)(ws);
    uint16_t* WcT  = (uint16_t*)(ws + szWaT);
    uint16_t* proj = (uint16_t*)(ws + szWaT + szWcT);
    uint16_t* hsT  = (uint16_t*)(ws + szWaT + szWcT + szProj);
    uint16_t* cbf  = (uint16_t*)(ws + szWaT + szWcT + szProj + szHsT);

    // 1) weight transpose/convert: (512*512 + 512*1024) elems
    wtrans_kernel<<<3072, 256, 0, stream>>>(Wa, Wc, WaT, WcT);
    // 2) hs -> hsT bf16 (K-contiguous per (i,n))
    hst_kernel<<<65536, 256, 0, stream>>>(hs, hsT);
    // 3) proj = hs @ Wa (bf16 WMMA)
    proj_gemm_kernel<<<dim3(2048, 2), 128, 0, stream>>>(hs, WaT, proj);
    // 4) fused score/softmax/context — 116 KB dynamic LDS (CDNA5 WGP has 320 KB)
    (void)hipFuncSetAttribute(reinterpret_cast<const void*>(&attn_kernel),
                              hipFuncAttributeMaxDynamicSharedMemorySize, kSmemSize);
    attn_kernel<<<dim3(kTT / 16, kB), 128, kSmemSize, stream>>>(ht, source, proj, hsT, cbf);
    // 5) out = tanh([c, ht] @ Wc + b)
    out_gemm_kernel<<<dim3(2048, 2), 128, 0, stream>>>(ht, cbf, WcT, bias, out);
}
